// ReCIT_45500883534634
// MI455X (gfx1250) — matrix operand order repaired and checked by reading, not compiled, not run on hardware
//
#include <hip/hip_runtime.h>
#include <hip/hip_bf16.h>
#include <math.h>

// ---------------- problem constants ----------------
#define NB   2
#define L    196
#define D    768
#define LP   208            // L padded to 13 WMMA tiles
#define LL   (L * L)        // 38416
#define LD   (L * D)        // 150528
#define NLL  (NB * LL)      // 76832
#define NLD  (NB * LD)      // 301056
#define LPL  (LP * L)       // 40768 (padded A-matrix size per batch)
#define LT   13             // LP/16 tiles along L
#define DT   48             // 768/16 tiles along D
#define TWO_PI 6.28318530717958647692f

typedef float        v2f  __attribute__((ext_vector_type(2)));
typedef float        v8f  __attribute__((ext_vector_type(8)));
typedef _Float16     v16h __attribute__((ext_vector_type(16)));
typedef unsigned int u32x4 __attribute__((ext_vector_type(4)));
typedef int          i32x4 __attribute__((ext_vector_type(4)));
typedef int          i32x8 __attribute__((ext_vector_type(8)));

#if defined(__has_builtin)
#if __has_builtin(__builtin_amdgcn_wmma_f32_16x16x4_f32)
#define USE_F32_WMMA 1
#else
#define USE_F32_WMMA 0
#endif
#if defined(__AMDGCN__) && __has_builtin(__builtin_amdgcn_tensor_load_to_lds) && \
    __has_builtin(__builtin_amdgcn_s_wait_tensorcnt)
#define HAVE_TDM 1
#else
#define HAVE_TDM 0
#endif
#else
#define USE_F32_WMMA 0
#define HAVE_TDM 0
#endif

// ---------------------------------------------------------------------------
// 16x16 output tile C = A(16xK) * B(Kx16) on the WMMA pipe (one wave, wave32).
// f32 path: V_WMMA_F32_16X16X4_F32 (exact f32; all our K are %4==0, no guards).
// A 16x4: lane = M (lane&15), lane>>4 selects K pair.  B 4x16: lane = N,
// VGPR = K row, lane>>4 selects K+2.  C/D: VGPR r holds M = r + 8*(lane>>4).
// ---------------------------------------------------------------------------
template <typename FA, typename FB>
__device__ __forceinline__ v8f wmma_tile(int K, FA loadA, FB loadB) {
  const int lane = threadIdx.x & 31;
  const int m    = lane & 15;
  v8f c;
  for (int r = 0; r < 8; ++r) c[r] = 0.0f;
#if USE_F32_WMMA
  const int kh = (lane >> 4) * 2;   // 0 or 2
  for (int k0 = 0; k0 < K; k0 += 4) {
    int ka = k0 + kh;
    v2f av, bv;
    av.x = loadA(m, ka);
    av.y = loadA(m, ka + 1);
    bv.x = loadB(ka,     m);
    bv.y = loadB(ka + 1, m);
    c = __builtin_amdgcn_wmma_f32_16x16x4_f32(false, av, false, bv,
                                              (short)0, c, false, false);
  }
#else
  const int hi = lane >> 4;
  for (int k0 = 0; k0 < K; k0 += 32) {
    v16h av, bv;
    for (int e = 0; e < 16; ++e) {
      int ka = k0 + e + 16 * hi;
      av[e] = (ka < K) ? (_Float16)loadA(m, ka) : (_Float16)0.0f;
      int kb = k0 + e + 16 * hi;
      bv[e] = (kb < K) ? (_Float16)loadB(kb, m) : (_Float16)0.0f;
    }
    c = __builtin_amdgcn_wmma_f32_16x16x32_f16(false, av, false, bv,
                                               (short)0, c, false, false);
  }
#endif
  return c;
}

// ---------------------------------------------------------------------------
// Tensor Data Mover: async 2D tile (tile_h rows x tile_w f32) from global
// (row stride = stride_elems) into contiguous LDS.  D# packed per ISA §8.3/8.4.
// tensor_dim == tile_dim so no OOB path is exercised (buffers are pre-padded).
// This toolchain's builtin is the 6-arg form: (g0, g1, g2, g3, g4, cpol);
// trailing groups are zero (count=1 descriptor, 2D tensor, no iterate/gather).
// ---------------------------------------------------------------------------
__device__ __forceinline__ void tdm_load_2d(void* lds, const void* gptr,
                                            unsigned tile_w, unsigned tile_h,
                                            unsigned stride_elems) {
#if HAVE_TDM
  unsigned lds_off = (unsigned)(unsigned long long)(uintptr_t)lds; // LDS byte offset
  unsigned long long ga = (unsigned long long)(uintptr_t)gptr;
  unsigned long long s0 = (unsigned long long)stride_elems;        // data_size units
  u32x4 g0;
  g0[0] = 1u;                                             // count=1 user D#
  g0[1] = lds_off;                                        // lds_addr [63:32]
  g0[2] = (unsigned)(ga & 0xFFFFFFFFu);                   // global_addr lo
  g0[3] = (unsigned)((ga >> 32) & 0x1FFFFFFu) | (2u << 30); // addr[56:32] | type=2
  i32x8 g1;
  g1[0] = (int)(2u << 16);                                // data_size=4B, mask=0
  g1[1] = (int)((tile_w & 0xFFFFu) << 16);                // tensor_dim0 lo16 @63:48
  g1[2] = (int)((tile_w >> 16) | ((tile_h & 0xFFFFu) << 16)); // dim0 hi16, dim1 lo16
  g1[3] = (int)((tile_h >> 16) | ((tile_w & 0xFFFFu) << 16)); // dim1 hi16, tile_dim0
  g1[4] = (int)(tile_h & 0xFFFFu);                        // tile_dim1, tile_dim2=0
  g1[5] = (int)(unsigned)(s0 & 0xFFFFFFFFu);              // dim0_stride lo32
  g1[6] = (int)(unsigned)((s0 >> 32) & 0xFFFFu);          // dim0_stride hi16
  g1[7] = 0;                                              // dim1_stride (unused, 2D)
  i32x4 gz4 = {0, 0, 0, 0};
  i32x8 gz8 = {0, 0, 0, 0, 0, 0, 0, 0};
  __builtin_amdgcn_tensor_load_to_lds(g0, g1, gz4, gz4, gz8, 0);
#else
  const int lane = threadIdx.x & 31;
  float* dst = (float*)lds;
  const float* src = (const float*)gptr;
  unsigned n = tile_w * tile_h;
  for (unsigned idx = lane; idx < n; idx += 32) {
    unsigned r = idx / tile_w, ccol = idx % tile_w;
    dst[idx] = src[(size_t)r * stride_elems + ccol];
  }
#endif
}

__device__ __forceinline__ void tdm_wait(const void* l0, const void* l1) {
#if HAVE_TDM
  __builtin_amdgcn_s_wait_tensorcnt(0);
#endif
  // LDS was written behind the compiler's back (TDM) — force it to reload.
  asm volatile("" : : "r"(l0), "r"(l1) : "memory");
}

// ---------------------------------------------------------------------------
// K1: patchify + standardize (ddof=1) + unit-norm rows -> hatP (rows >=196 zero)
// ---------------------------------------------------------------------------
__global__ void k_patchify(const float* __restrict__ imgs,
                           float* __restrict__ hat) {
  const int a = blockIdx.x / LP;
  const int l = blockIdx.x % LP;
  const int tid = threadIdx.x;
  if (l >= L) {                       // zero pad rows (uniform per block)
    for (int e = 0; e < 3; ++e)
      hat[((size_t)a * LP + l) * D + tid + 256 * e] = 0.0f;
    return;
  }
  const int ph = l / 14, pw = l % 14;
  __shared__ float red[256];
  float v[3], s1 = 0.0f, s2 = 0.0f;
  for (int e = 0; e < 3; ++e) {
    int d = tid + 256 * e;
    int p = d / 48, q = (d % 48) / 3, ch = d % 3;
    float x = imgs[((a * 3 + ch) * 224 + (ph * 16 + p)) * 224 + (pw * 16 + q)];
    v[e] = x; s1 += x; s2 += x * x;
  }
  red[tid] = s1; __syncthreads();
  for (int s = 128; s > 0; s >>= 1) { if (tid < s) red[tid] += red[tid + s]; __syncthreads(); }
  s1 = red[0]; __syncthreads();
  red[tid] = s2; __syncthreads();
  for (int s = 128; s > 0; s >>= 1) { if (tid < s) red[tid] += red[tid + s]; __syncthreads(); }
  s2 = red[0];
  float mean = s1 / 768.0f;
  float ssd  = fmaxf(s2 - s1 * s1 / 768.0f, 0.0f);
  float var  = ssd / 767.0f;
  float inv  = rsqrtf(var + 1e-6f);
  float nrm  = sqrtf(ssd) * inv;
  float scale = inv / fmaxf(nrm, 1e-20f);
  for (int e = 0; e < 3; ++e) {
    int d = tid + 256 * e;
    hat[((size_t)a * LP + l) * D + d] = (v[e] - mean) * scale;
  }
}

// ---------------------------------------------------------------------------
// K2: Sim = hatP @ hatP^T, M0 = triu(Sim > 0.3) into padded MfP.  1 wave/tile.
// Unconditional b64 loads (rows padded); pad rows of hatP are zero.
// ---------------------------------------------------------------------------
__global__ void k_sim(const float* __restrict__ hat, float* __restrict__ Mf) {
  const int blk = blockIdx.x;
  const int a  = blk / (LT * LT);
  const int ti = (blk / LT) % LT;
  const int tj = blk % LT;
  const int I = ti * 16, J = tj * 16;
  const float* A = hat + (size_t)a * LP * D;
  auto loadA = [&](int m, int k) -> float { return A[(size_t)(I + m) * D + k]; };
  auto loadB = [&](int k, int n) -> float { return A[(size_t)(J + n) * D + k]; };
  v8f c = wmma_tile(D, loadA, loadB);
  const int lane = threadIdx.x & 31;
  const int n  = lane & 15;
  const int mb = (lane >> 4) * 8;
  for (int r = 0; r < 8; ++r) {
    int row = I + mb + r, col = J + n;
    if (row < L && col < L)
      Mf[(size_t)a * LPL + row * L + col] =
          (c[r] > 0.3f && col >= row) ? 1.0f : 0.0f;
  }
}

// ---------------------------------------------------------------------------
// K3: sequential masking loop; whole M (300 KiB) in one WGP's 320 KB LDS.
// Writes padded MfP (pad rows zero) + per-row counts num.
// ---------------------------------------------------------------------------
__global__ void k_mloop(float* __restrict__ Mf, float* __restrict__ numv) {
  extern __shared__ float smem[];
  float* Ms  = smem;          // NLL floats
  float* red = smem + NLL;    // 1024 floats
  const int tid = threadIdx.x;
  for (int i = tid; i < NLL; i += 1024) {
    int a = i / LL;
    Ms[i] = Mf[(size_t)a * LPL + (i % LL)];
  }
  __syncthreads();
  for (int i = 0; i < L - 1; ++i) {
    float part = 0.0f;
    for (int idx = tid; idx < NLL; idx += 1024) part += Ms[idx];
    red[tid] = part; __syncthreads();
    for (int s = 512; s > 0; s >>= 1) { if (tid < s) red[tid] += red[tid + s]; __syncthreads(); }
    float total = red[0];
    __syncthreads();
    if (total != (float)(NB * L)) {   // reference freezes M once sum == N*L
      for (int idx = tid; idx < NLL; idx += 1024) {
        int k = idx % L;
        int j = (idx / L) % L;
        int a = idx / LL;
        if (j > i && Ms[a * LL + i * L + k] != 0.0f) Ms[idx] = 0.0f;
      }
    }
    __syncthreads();
  }
  for (int idx = tid; idx < NB * LPL; idx += 1024) {
    int k   = idx % L;
    int row = (idx / L) % LP;
    int a   = idx / LPL;
    Mf[idx] = (row < L) ? Ms[a * LL + row * L + k] : 0.0f;
  }
  if (tid < NB * L) {
    int a = tid / L, i = tid % L;
    float s = 0.0f;
    for (int k = 0; k < L; ++k) s += Ms[a * LL + i * L + k];
    numv[tid] = s;
  }
}

// ---------------------------------------------------------------------------
// K4: 768-point DFT per token (exact integer phase (k*t) mod 768, LDS tables)
// ---------------------------------------------------------------------------
__global__ void k_dft(const float* __restrict__ x, float* __restrict__ sty,
                      float* __restrict__ sty2, float* __restrict__ con) {
  const int a = blockIdx.x / L;
  const int l = blockIdx.x % L;
  const int tid = threadIdx.x;
  __shared__ float xs[D], tc[D], ts[D];
  for (int e = 0; e < 3; ++e) {
    int d = tid + 256 * e;
    xs[d] = x[((size_t)a * (L + 1) + 1 + l) * D + d];
    float ang = (TWO_PI / 768.0f) * (float)d;
    tc[d] = cosf(ang);
    ts[d] = sinf(ang);
  }
  __syncthreads();
  for (int e = 0; e < 3; ++e) {
    int k = tid + 256 * e;
    float re = 0.0f, im = 0.0f;
    int m = 0;
    for (int t = 0; t < D; ++t) {
      float xv = xs[t];
      re += xv * tc[m];
      im -= xv * ts[m];
      m += k; if (m >= D) m -= D;
    }
    float s = sqrtf(re * re + im * im);
    size_t o = ((size_t)a * L + l) * D + k;
    sty[o]  = s;
    sty2[o] = s * s;
    con[o]  = atan2f(im, re);
  }
}

// K5a: padded copies of learnable_sty and learnable_sty*noise (pad rows zero)
__global__ void k_padA(const float* __restrict__ lsty,
                       const float* __restrict__ nois,
                       float* __restrict__ lstyP, float* __restrict__ AnP) {
  int idx = blockIdx.x * 256 + threadIdx.x;
  if (idx >= NB * LPL) return;
  int k   = idx % L;
  int row = (idx / L) % LP;
  int a   = idx / LPL;
  float lv = 0.0f, av = 0.0f;
  if (row < L) {
    size_t o = (size_t)a * LL + row * L + k;
    lv = lsty[o];
    av = lv * nois[o];
  }
  lstyP[idx] = lv;
  AnP[idx]   = av;
}

// ---------------------------------------------------------------------------
// K5: C[a] = A[a](LP x L, padded) @ B[a](L x D).  TDM stages the A tile
// (16 x 196) and B tile (196 x 16, stride D) into LDS; s_wait_tensorcnt 0;
// then the f32 WMMA K-loop reads LDS (ds_load_b64).  One wave per 16x16 tile.
// ---------------------------------------------------------------------------
__global__ void k_gemm(const float* __restrict__ A, const float* __restrict__ B,
                       float* __restrict__ Cmat) {
  __shared__ float ldsA[16 * L];   // 12544 B
  __shared__ float ldsB[L * 16];   // 12544 B
  const int blk = blockIdx.x;
  const int a  = blk / (LT * DT);
  const int ti = (blk / DT) % LT;
  const int tj = blk % DT;
  const int I = ti * 16, J = tj * 16;
  const float* Atile = A + (size_t)a * LPL + (size_t)I * L;   // 16 rows x 196
  const float* Btile = B + (size_t)a * LD + J;                // 196 rows x 16
  tdm_load_2d(ldsA, Atile, /*w=*/L,  /*h=*/16, /*stride=*/L);
  tdm_load_2d(ldsB, Btile, /*w=*/16, /*h=*/L,  /*stride=*/D);
  tdm_wait(ldsA, ldsB);
  auto loadA = [&](int m, int k) -> float { return ldsA[m * L + k]; };
  auto loadB = [&](int k, int n) -> float { return ldsB[k * 16 + n]; };
  v8f c = wmma_tile(L, loadA, loadB);
  const int lane = threadIdx.x & 31;
  const int n  = lane & 15;
  const int mb = (lane >> 4) * 8;
  for (int r = 0; r < 8; ++r) {
    int row = I + mb + r;
    if (row < L) Cmat[(size_t)a * LD + (size_t)row * D + J + n] = c[r];
  }
}

// K6: avg / std / mask products (elementwise over N*L*D)
__global__ void k_stats(const float* __restrict__ avgs,
                        const float* __restrict__ sumsq,
                        const float* __restrict__ numv,
                        float* __restrict__ mko, float* __restrict__ mks,
                        float* __restrict__ mka) {
  int idx = blockIdx.x * 256 + threadIdx.x;
  if (idx >= NLD) return;
  int ai = idx / D;
  float num  = numv[ai];
  float numc = fmaxf(num, 1e-7f);
  float asum = avgs[idx];
  float avg  = asum / numc;
  float varn = (sumsq[idx] - 2.0f * avg * asum + num * avg * avg) / numc;
  float sd   = sqrtf(fmaxf(varn, 0.0f));
  float mk   = (asum > 0.0f) ? 1.0f : 0.0f;
  mko[idx] = mk;
  mks[idx] = mk * sd;
  mka[idx] = mk * avg;
}

// K7: sum_sty = (T1 + T2) / clip(Z, 1e-7)
__global__ void k_combine(const float* __restrict__ T1,
                          const float* __restrict__ T2,
                          const float* __restrict__ Zb,
                          float* __restrict__ ssty) {
  int idx = blockIdx.x * 256 + threadIdx.x;
  if (idx < NLD) ssty[idx] = (T1[idx] + T2[idx]) / fmaxf(Zb[idx], 1e-7f);
}

// ---------------------------------------------------------------------------
// K8: inverse DFT of sum_sty * e^{i con} (real part) + cls-token passthrough
// ---------------------------------------------------------------------------
__global__ void k_idft(const float* __restrict__ x, const float* __restrict__ ss,
                       const float* __restrict__ con, float* __restrict__ out) {
  const int a = blockIdx.x / (L + 1);
  const int l = blockIdx.x % (L + 1);
  const int tid = threadIdx.x;
  if (l == 0) {
    for (int e = 0; e < 3; ++e) {
      int d = tid + 256 * e;
      out[(size_t)(a * (L + 1)) * D + d] = x[(size_t)(a * (L + 1)) * D + d];
    }
    return;
  }
  const int li = l - 1;
  __shared__ float cr[D], ci[D], tc[D], ts[D];
  for (int e = 0; e < 3; ++e) {
    int k = tid + 256 * e;
    size_t o = ((size_t)a * L + li) * D + k;
    float m = ss[o], th = con[o];
    cr[k] = m * cosf(th);
    ci[k] = m * sinf(th);
    float ang = (TWO_PI / 768.0f) * (float)k;
    tc[k] = cosf(ang);
    ts[k] = sinf(ang);
  }
  __syncthreads();
  for (int e = 0; e < 3; ++e) {
    int t = tid + 256 * e;
    float acc = 0.0f;
    int m = 0;
    for (int k = 0; k < D; ++k) {
      acc += cr[k] * tc[m] - ci[k] * ts[m];
      m += t; if (m >= D) m -= D;
    }
    out[((size_t)a * (L + 1) + l) * D + t] = acc / 768.0f;
  }
}

// ---------------------------------------------------------------------------
extern "C" void kernel_launch(void* const* d_in, const int* in_sizes, int n_in,
                              void* d_out, int out_size, void* d_ws, size_t ws_size,
                              hipStream_t stream) {
  (void)in_sizes; (void)n_in; (void)out_size; (void)ws_size;
  const float* x    = (const float*)d_in[0];   // (2,197,768)
  const float* imgs = (const float*)d_in[1];   // (2,3,224,224)
  const float* lsty = (const float*)d_in[2];   // (2,196,196,1)
  const float* nois = (const float*)d_in[3];   // (2,196,196,1)
  float* out = (float*)d_out;                  // (2,197,768)

  float* w = (float*)d_ws;
  float* hatP  = w; w += (size_t)NB * LP * D;  // padded rows
  float* sty   = w; w += NLD;
  float* sty2  = w; w += NLD;
  float* con   = w; w += NLD;
  float* avgs  = w; w += NLD;
  float* sumsq = w; w += NLD;
  float* mko   = w; w += NLD;
  float* mks   = w; w += NLD;
  float* mka   = w; w += NLD;
  float* Zb    = w; w += NLD;
  float* T1    = w; w += NLD;
  float* T2    = w; w += NLD;
  float* ssty  = w; w += NLD;
  float* MfP   = w; w += (size_t)NB * LPL;     // padded rows
  float* lstyP = w; w += (size_t)NB * LPL;
  float* AnP   = w; w += (size_t)NB * LPL;
  float* numv  = w; w += NB * L;

  k_patchify<<<NB * LP, 256, 0, stream>>>(imgs, hatP);
  k_sim<<<NB * LT * LT, 32, 0, stream>>>(hatP, MfP);
  size_t smbytes = (size_t)NLL * sizeof(float) + 1024 * sizeof(float); // ~304 KiB
  k_mloop<<<1, 1024, smbytes, stream>>>(MfP, numv);
  k_dft<<<NB * L, 256, 0, stream>>>(x, sty, sty2, con);
  k_padA<<<(NB * LPL + 255) / 256, 256, 0, stream>>>(lsty, nois, lstyP, AnP);
  k_gemm<<<NB * LT * DT, 32, 0, stream>>>(MfP, sty, avgs);    // avg_sum = M @ sty
  k_gemm<<<NB * LT * DT, 32, 0, stream>>>(MfP, sty2, sumsq);  // sumsq   = M @ sty^2
  k_stats<<<(NLD + 255) / 256, 256, 0, stream>>>(avgs, sumsq, numv, mko, mks, mka);
  k_gemm<<<NB * LT * DT, 32, 0, stream>>>(lstyP, mko, Zb);    // Z  = lsty @ mask
  k_gemm<<<NB * LT * DT, 32, 0, stream>>>(AnP,   mks, T1);    // T1 = (lsty*noise) @ (mask*std)
  k_gemm<<<NB * LT * DT, 32, 0, stream>>>(lstyP, mka, T2);    // T2 = lsty @ (mask*avg)
  k_combine<<<(NLD + 255) / 256, 256, 0, stream>>>(T1, T2, Zb, ssty);
  k_idft<<<NB * (L + 1), 256, 0, stream>>>(x, ssty, con, out);
}
